// MSDNet_8340826489537
// MI455X (gfx1250) — matrix operand order repaired and checked by reading, not compiled, not run on hardware
//
#include <hip/hip_runtime.h>
#include <hip/hip_bf16.h>

// MSDNet on gfx1250 (MI455X), fully L2-resident recurrence (128 MiB workspace < 192 MB L2):
//   feats_h: f16, pixel-major [NPIX][32]          (64 MiB)  channel c of pixel q at feats_h[q*32+c]
//   Y:       f32, blocked  [NPIX/16][16 taps][16] (64 MiB)  1 KB per 16-pixel chunk
//   Per layer i (n_in = i+1, dil = (i%10)+1):
//     stage1 (WMMA): D[pix, tap] = sum_c X[pix, c] * W[c, tap]
//        A = X chunk (16 px x 32 chan, two b128 loads/lane, software-pipelined)
//        B = weight tile (32 chan x 16 taps, built once per wave, zero-padded)
//        one v_wmma_f32_16x16x32_f16 per 16 pixels; D stored as one aligned v8f per lane,
//        branchless, uniform across the wave.
//     stage2 (VALU): feats[q, 1+i] = relu(bias_i + sum_t Y[t at q + dil*off_t])  (zero pad)
//   Final: 1x1 conv over 31 channels (f16 in, fp32 accum) + bias + output affine.
// Zero-padded channel slots + zero weight columns make every load/store branchless.

typedef __attribute__((ext_vector_type(16))) _Float16 v16h;
typedef __attribute__((ext_vector_type(8)))  _Float16 v8h;
typedef __attribute__((ext_vector_type(8)))  float    v8f;

#define BATCH 4
#define HDIM  512
#define WDIM  512
#define HW    (HDIM * WDIM)
#define NPIX  (BATCH * HW)       // 1,048,576
#define DEPTH 30
#define CTOT  (DEPTH + 1)        // 31 real channels, padded to 32 slots
#define CPAD  32
#define YBLK  256                // floats per 16-pixel Y block: 16 taps x 16 px

// ---------------- init: pixel block = {x*sin_w+sin_b, 0, ..., 0} ----------------
__global__ void msd_init(const float* __restrict__ x,
                         const float* __restrict__ sin_w,
                         const float* __restrict__ sin_b,
                         _Float16* __restrict__ feats) {
    int q = blockIdx.x * blockDim.x + threadIdx.x;      // 0..NPIX-1
    v16h lo = {};
    v16h hi = {};
    lo[0] = (_Float16)(x[q] * sin_w[0] + sin_b[0]);
    v16h* dst = (v16h*)(feats + (size_t)q * CPAD);
    dst[0] = lo;                                        // channels 0..15
    dst[1] = hi;                                        // channels 16..31 (zero)
}

// ---------------- stage1: pointwise tap-GEMM via WMMA (A = pixels, B = weights) ----------------
__global__ void msd_stage1(const _Float16* __restrict__ feats,
                           const float* __restrict__ Wmsd,
                           float* __restrict__ Y,
                           int layer) {
    const int n_in  = layer + 1;                 // channels feeding this layer (<= 30)
    const int lane  = threadIdx.x & 31;
    const int nlo   = lane & 15;                 // A row (pixel) / B,D column (tap)
    const int khalf = lane >> 4;

    // B tile = weights (32 chan x 16 taps), resident for the whole kernel.
    // B layout: lane = column (tap), element e<8 -> K = khalf*8 + e, e>=8 -> K = 16 + khalf*8 + e-8. Branchless clamped build.
    v16h bw;
    {
        const int t = (nlo < 9) ? nlo : 8;
#pragma unroll
        for (int e = 0; e < 16; ++e) {
            int   k  = (e < 8) ? (khalf * 8 + e) : (16 + khalf * 8 + (e - 8));
            int   kc = (k < n_in) ? k : (n_in - 1);
            float w  = Wmsd[((size_t)layer * DEPTH + kc) * 9 + t];
            w = (k < n_in && nlo < 9) ? w : 0.0f;
            bw[e] = (_Float16)w;
        }
    }

    const int    wave = (blockIdx.x * blockDim.x + threadIdx.x) >> 5;  // 0..4095
    const size_t q0   = (size_t)wave * 256;      // 16 chunks of 16 pixels per wave

    // A chunk loader. A layout (16x32 f16): lane = row (pixel), elements
    //   e<8  -> K = khalf*8 + e        (bytes [khalf*16,      +16) of the pixel block)
    //   e>=8 -> K = 16 + khalf*8 + e-8 (bytes [32 + khalf*16, +16))
    auto loadA = [&](size_t qbase) -> v16h {
        const _Float16* blk = feats + (qbase + (size_t)nlo) * CPAD + khalf * 8;
        v8h lo = *(const v8h*)(blk);
        v8h hi = *(const v8h*)(blk + 16);
        return __builtin_shufflevector(lo, hi, 0, 1, 2, 3, 4, 5, 6, 7,
                                       8, 9, 10, 11, 12, 13, 14, 15);
    };

    v16h aCur = loadA(q0);                       // software pipeline: prime first chunk
    for (int ch = 0; ch < 16; ++ch) {
        const size_t qbase = q0 + (size_t)ch * 16;

        // Prefetch chunk+2, load chunk+1 BEFORE the WMMA on the current chunk,
        // so the next s_wait_loadcnt overlaps this WMMA + stores.
        __builtin_prefetch((const void*)(feats + (qbase + 32 + nlo) * CPAD), 0, 1);
        v16h aNext = aCur;
        if (ch + 1 < 16) aNext = loadA(qbase + 16);

        v8f c = {};
        // All 32 lanes active at the WMMA.
        c = __builtin_amdgcn_wmma_f32_16x16x32_f16(false, aCur, false, bw,
                                                   (short)0, c, false, false);

        // D: lane = tap column nlo; c[0..7] = pixels (8*khalf)..(8*khalf+7).
        // One aligned 32B store per lane, uniform across the wave (taps 9..15 store zeros).
        float* dst = Y + (qbase >> 4) * YBLK + (size_t)nlo * 16 + (size_t)khalf * 8;
        *(v8f*)dst = c;

        aCur = aNext;
    }
}

// ---------------- stage2: dilated shift-add + bias + relu -> feats channel (1+i) ----------------
__global__ void msd_stage2(const float* __restrict__ Y,
                           const float* __restrict__ bias,
                           _Float16* __restrict__ feats,
                           int layer, int dil) {
    int q = blockIdx.x * blockDim.x + threadIdx.x;
    int b = q / HW;
    int p = q - b * HW;
    int y = p / WDIM;
    int x = p - y * WDIM;

    float acc = bias[layer];
#pragma unroll
    for (int ty = 0; ty < 3; ++ty) {
#pragma unroll
        for (int tx = 0; tx < 3; ++tx) {
            int yy = y + dil * (ty - 1);
            int xx = x + dil * (tx - 1);
            if (yy >= 0 && yy < HDIM && xx >= 0 && xx < WDIM) {
                int s = b * HW + yy * WDIM + xx;          // shifted global pixel
                acc += Y[(size_t)(s >> 4) * YBLK + (ty * 3 + tx) * 16 + (s & 15)];
            }
        }
    }
    feats[(size_t)q * CPAD + (layer + 1)] = (_Float16)fmaxf(acc, 0.0f);
}

// ---------------- final: 1x1 conv over 31 channels + output affine ----------------
__global__ void msd_final(const _Float16* __restrict__ feats,
                          const float* __restrict__ convW,
                          const float* __restrict__ convB,
                          const float* __restrict__ sout_w,
                          const float* __restrict__ sout_b,
                          float* __restrict__ out) {
    int q = blockIdx.x * blockDim.x + threadIdx.x;

    const _Float16* fb = feats + (size_t)q * CPAD;        // contiguous 62B per thread
    float acc = 0.0f;
#pragma unroll
    for (int c = 0; c < CTOT; ++c)
        acc += convW[c] * (float)fb[c];
    out[q] = (acc + convB[0]) * sout_w[0] + sout_b[0];
}

extern "C" void kernel_launch(void* const* d_in, const int* in_sizes, int n_in,
                              void* d_out, int out_size, void* d_ws, size_t ws_size,
                              hipStream_t stream) {
    const float* x      = (const float*)d_in[0];   // [4,1,512,512]
    const float* Wmsd   = (const float*)d_in[1];   // [30,30,3,3]
    const float* bias   = (const float*)d_in[2];   // [30]
    const float* convW  = (const float*)d_in[3];   // [1,31,1,1]
    const float* convB  = (const float*)d_in[4];   // [1]
    const float* sin_w  = (const float*)d_in[5];
    const float* sin_b  = (const float*)d_in[6];
    const float* sout_w = (const float*)d_in[7];
    const float* sout_b = (const float*)d_in[8];
    float*       out    = (float*)d_out;

    // Workspace: feats_h = NPIX*32*2B = 64 MiB, then Y = (NPIX/16)*256*4B = 64 MiB (128 MiB total)
    _Float16* feats = (_Float16*)d_ws;
    float*    Y     = (float*)((char*)d_ws + (size_t)NPIX * CPAD * sizeof(_Float16));

    const int elemBlocks = NPIX / 256;               // 4096
    const int s1Blocks   = (NPIX / 256) * 32 / 256;  // 4096 waves -> 512 blocks of 8 waves

    msd_init<<<elemBlocks, 256, 0, stream>>>(x, sin_w, sin_b, feats);

    for (int i = 0; i < DEPTH; ++i) {
        const int dil = (i % 10) + 1;
        msd_stage1<<<s1Blocks, 256, 0, stream>>>(feats, Wmsd, Y, i);
        msd_stage2<<<elemBlocks, 256, 0, stream>>>(Y, bias, feats, i, dil);
    }

    msd_final<<<elemBlocks, 256, 0, stream>>>(feats, convW, convB, sout_w, sout_b, out);
}
